// GCN_75917841924646
// MI455X (gfx1250) — matrix operand order repaired and checked by reading, not compiled, not run on hardware
//
#include <hip/hip_runtime.h>
#include <hip/hip_bf16.h>

#define N_NODES 100000
#define N_EDGES 1600000
#define NFEAT 128
#define NHID 128
#define NCLASS 64

typedef __attribute__((ext_vector_type(16))) __bf16 v16bf;
typedef __attribute__((ext_vector_type(8)))  float  v8f;

union FragAB { uint4 u[2]; v16bf v; };

// ---------------------------------------------------------------- zero
__global__ __launch_bounds__(256) void zero_kernel(float* __restrict__ p, long long n) {
    long long i = (long long)blockIdx.x * blockDim.x + threadIdx.x;
    long long stride = (long long)gridDim.x * blockDim.x;
    for (; i < n; i += stride) p[i] = 0.0f;
}

// ---------------------------------------------------------------- SpMM (edge-parallel, wave32 per edge)
// h: [N,128] f32 viewed as 32 float4 per row. Each lane handles 4 features.
__global__ __launch_bounds__(256) void spmm_kernel(const float4* __restrict__ h,
                                                   const int* __restrict__ src,
                                                   const int* __restrict__ dst,
                                                   const float* __restrict__ w,
                                                   float* __restrict__ out,
                                                   int n_edges) {
    int gid  = blockIdx.x * blockDim.x + threadIdx.x;
    int e    = gid >> 5;          // one wave32 per edge
    int lane = gid & 31;
    if (e >= n_edges) return;
    int   s  = src[e];
    int   d  = dst[e];
    float ww = w[e];
    float4 v = h[(long long)s * 32 + lane];          // coalesced 512B per wave, L2-resident
    float* o = out + (long long)d * 128 + lane * 4;
    atomicAdd(o + 0, v.x * ww);
    atomicAdd(o + 1, v.y * ww);
    atomicAdd(o + 2, v.z * ww);
    atomicAdd(o + 3, v.w * ww);
}

// ---------------------------------------------------------------- f32 -> bf16 convert
__global__ __launch_bounds__(256) void cvt_bf16_kernel(const float* __restrict__ in,
                                                       __hip_bfloat16* __restrict__ out,
                                                       long long n) {
    long long i = (long long)blockIdx.x * blockDim.x + threadIdx.x;
    long long stride = (long long)gridDim.x * blockDim.x;
    for (; i < n; i += stride) out[i] = __float2bfloat16(in[i]);
}

// ---------------------------------------------------------------- WMMA GEMM: C[M, NT*16] = A[M,128] @ W[128, NT*16] + bias (+ReLU)
// A is bf16 row-major. W/bias f32; W is packed to bf16 B-fragment layout in LDS once per block.
// One wave per 16-row tile; K = 128 = 4 steps of 32; NT column tiles of 16.
template<int NT, bool RELU>
__global__ __launch_bounds__(256) void gemm_wmma_kernel(const __hip_bfloat16* __restrict__ A,
                                                        const float* __restrict__ W,
                                                        const float* __restrict__ bias,
                                                        float* __restrict__ C,
                                                        int M) {
    constexpr int NCOL = NT * 16;
    __shared__ __align__(32) __hip_bfloat16 ldsW[4 * NT * 32 * 16]; // [kt][nt][lane][16] = 128*NCOL bf16

    int tid = threadIdx.x;
    // Pack W into B-fragment order: lane L of tile (kt,nt) holds W[kt*32 + (i>>3)*16 + (L>>4)*8 + (i&7)][nt*16 + (L&15)], i=0..15.
    for (int idx = tid; idx < 128 * NCOL; idx += 256) {
        int j    = idx & 15;
        int lane = (idx >> 4) & 31;
        int t    = idx >> 9;            // kt*NT + nt
        int nt   = t % NT;
        int kt   = t / NT;
        int K    = kt * 32 + (j >> 3) * 16 + (lane >> 4) * 8 + (j & 7);
        int N    = nt * 16 + (lane & 15);
        ldsW[idx] = __float2bfloat16(W[K * NCOL + N]);
    }
    __syncthreads();

    int wave    = tid >> 5;
    int lane    = tid & 31;
    int rowTile = blockIdx.x * 8 + wave;
    if (rowTile * 16 >= M) return;

    int h   = lane >> 4;                 // half-wave selector
    int row = rowTile * 16 + (lane & 15);

    v8f acc[NT] = {};

    const __hip_bfloat16* arow = A + (long long)row * 128;
#pragma unroll
    for (int kt = 0; kt < 4; ++kt) {
        // A fragment (16x32 bf16): a[0..7]=K kt*32+h*8+0..7 ; a[8..15]=K kt*32+16+h*8+0..7  (two 16B loads)
        FragAB a;
        a.u[0] = *reinterpret_cast<const uint4*>(arow + kt * 32 + h * 8);
        a.u[1] = *reinterpret_cast<const uint4*>(arow + kt * 32 + 16 + h * 8);
#pragma unroll
        for (int nt = 0; nt < NT; ++nt) {
            v16bf b = *reinterpret_cast<const v16bf*>(&ldsW[((kt * NT + nt) * 32 + lane) * 16]);
            acc[nt] = __builtin_amdgcn_wmma_f32_16x16x32_bf16(
                false, a.v, false, b, (short)0, acc[nt], false, false);
        }
    }

    // D layout: VGPR j -> row h*8 + j ; col = lane&15 within tile nt
    int baseRow = rowTile * 16 + h * 8;
#pragma unroll
    for (int nt = 0; nt < NT; ++nt) {
        int   col = nt * 16 + (lane & 15);
        float bv  = bias[col];
#pragma unroll
        for (int j = 0; j < 8; ++j) {
            float val = acc[nt][j] + bv;
            if (RELU) val = fmaxf(val, 0.0f);
            C[(long long)(baseRow + j) * NCOL + col] = val;
        }
    }
}

// ---------------------------------------------------------------- log_softmax over 64 classes, one wave32 per row
__global__ __launch_bounds__(256) void logsoftmax_kernel(const float* __restrict__ logits,
                                                         float* __restrict__ out, int M) {
    int gid  = blockIdx.x * blockDim.x + threadIdx.x;
    int r    = gid >> 5;
    int lane = gid & 31;
    if (r >= M) return;
    float2 v = reinterpret_cast<const float2*>(logits + (long long)r * 64)[lane];
    float m = fmaxf(v.x, v.y);
#pragma unroll
    for (int off = 16; off > 0; off >>= 1) m = fmaxf(m, __shfl_xor(m, off, 32));
    float s = expf(v.x - m) + expf(v.y - m);
#pragma unroll
    for (int off = 16; off > 0; off >>= 1) s += __shfl_xor(s, off, 32);
    float lse = m + logf(s);
    float2 o; o.x = v.x - lse; o.y = v.y - lse;
    reinterpret_cast<float2*>(out + (long long)r * 64)[lane] = o;
}

// ----------------------------------------------------------------
extern "C" void kernel_launch(void* const* d_in, const int* in_sizes, int n_in,
                              void* d_out, int out_size, void* d_ws, size_t ws_size,
                              hipStream_t stream) {
    (void)in_sizes; (void)n_in; (void)out_size; (void)ws_size;
    const float* x    = (const float*)d_in[0];
    const int*   esrc = (const int*)d_in[1];
    const int*   edst = (const int*)d_in[2];
    const float* ew   = (const float*)d_in[3];
    const float* W0   = (const float*)d_in[4];
    const float* b0   = (const float*)d_in[5];
    const float* W1   = (const float*)d_in[6];
    const float* b1   = (const float*)d_in[7];
    float* out = (float*)d_out;

    // Workspace layout (all offsets 256B aligned):
    //  B0: f32  [N,128]  51,200,000 B  (aggregation buffer, later reused for logits [N,64])
    //  B1: bf16 [N,128]  25,600,000 B  (bf16 copy of aggregated features -> WMMA A operand)
    //  B2: f32  [N,128]  51,200,000 B  (hidden layer output)
    char* ws = (char*)d_ws;
    float*          B0 = (float*)ws;
    __hip_bfloat16* B1 = (__hip_bfloat16*)(ws + 51200000);
    float*          B2 = (float*)(ws + 51200000 + 25600000);

    const long long nElem      = (long long)N_NODES * 128;
    const int       spmmBlocks = (N_EDGES * 32) / 256;           // 200000
    const int       gemmBlocks = (N_NODES / 16 + 7) / 8;         // 782

    // Layer 1: aggregate -> GEMM(+bias,ReLU)
    zero_kernel<<<2048, 256, 0, stream>>>(B0, nElem);
    spmm_kernel<<<spmmBlocks, 256, 0, stream>>>((const float4*)x, esrc, edst, ew, B0, N_EDGES);
    cvt_bf16_kernel<<<2048, 256, 0, stream>>>(B0, B1, nElem);
    gemm_wmma_kernel<8, true><<<gemmBlocks, 256, 0, stream>>>(B1, W0, b0, B2, N_NODES);

    // Layer 2: aggregate -> GEMM(+bias) -> log_softmax
    zero_kernel<<<2048, 256, 0, stream>>>(B0, nElem);
    spmm_kernel<<<spmmBlocks, 256, 0, stream>>>((const float4*)B2, esrc, edst, ew, B0, N_EDGES);
    cvt_bf16_kernel<<<2048, 256, 0, stream>>>(B0, B1, nElem);
    gemm_wmma_kernel<4, false><<<gemmBlocks, 256, 0, stream>>>(B1, W1, b1, B0, N_NODES);

    logsoftmax_kernel<<<((long long)N_NODES * 32 + 255) / 256, 256, 0, stream>>>(B0, out, N_NODES);
}
